// GeneGCN_5342939316652
// MI455X (gfx1250) — matrix operand order repaired and checked by reading, not compiled, not run on hardware
//
#include <hip/hip_runtime.h>
#include <hip/hip_bf16.h>

// GCN 2-layer: h1 = relu(A @ (X W1)); out = relu(A @ (h1 W2))
// N=100000 nodes, E=1600000 edges, all feature dims = 128.
//
// Pipeline (everything on `stream`, graph-capture safe):
//   1. convert W1,W2 f32 -> bf16 (into ws)
//   2. GEMM1: X(f32, cvt bf16 inline) @ W1bf -> bufA        [v_wmma_f32_16x16x32_bf16]
//      (W staged into LDS per workgroup via TENSOR_LOAD_TO_LDS / TDM)
//   3. memset d_out = 0 ; SPMM1: edge atomics into d_out    [global_atomic_add_f32]
//   4. GEMM2: relu(d_out) (cvt bf16 inline) @ W2bf -> bufA  (relu fused into A-load)
//   5. memset d_out = 0 ; SPMM2: edge atomics into d_out
//   6. relu(d_out) in place
//
// Workspace layout: [0,32KB) W1bf | [32KB,64KB) W2bf | [64KB, 64KB+51.2MB) bufA f32
// Required ws_size: 64*1024 + 100000*128*4 bytes (~51.3 MB).

#define N_NODES 100000
#define N_EDGES 1600000
#define DIM     128
#define N_TILES (N_NODES / 16)   // 6250, exact

typedef __attribute__((ext_vector_type(16))) __bf16       v16bf;
typedef __attribute__((ext_vector_type(8)))  float        v8f;
typedef __attribute__((ext_vector_type(4)))  unsigned int v4u;
typedef __attribute__((ext_vector_type(8)))  int          v8i;
typedef __attribute__((ext_vector_type(4)))  int          v4i;

#if __has_builtin(__builtin_amdgcn_tensor_load_to_lds)
#define GCN_USE_TDM 1
#else
#define GCN_USE_TDM 0
#endif

// ---------------------------------------------------------------------------
// W f32 -> bf16 conversion (both weight matrices; 128x128 each)
// ---------------------------------------------------------------------------
__global__ __launch_bounds__(256) void gcn_convert_w(
    const float* __restrict__ W1, const float* __restrict__ W2,
    __bf16* __restrict__ W1b, __bf16* __restrict__ W2b)
{
    int idx = blockIdx.x * blockDim.x + threadIdx.x;
    if (idx < DIM * DIM) {
        W1b[idx] = (__bf16)W1[idx];
        W2b[idx] = (__bf16)W2[idx];
    }
}

// ---------------------------------------------------------------------------
// GEMM: H[N,128] = act(A[N,128]) @ Wb[128,128]   (bf16 WMMA, f32 accum)
// One wave per 16-row tile; wave computes 16x128 via 8 accumulators,
// looping K in 4 chunks of 32. The 32KB bf16 weight tile is DMA'd into LDS
// once per workgroup by the Tensor Data Mover (TENSORcnt-tracked), then the
// hot loop reads B fragments with strided 16-bit LDS loads.
//
// Fragment layouts per CDNA5 ISA 7.12.2 (wave32):
//  A (16x32 bf16): lane l -> row m = l&15, half = l>>4.
//     a[0..7]  = K 32kt+8h+0..7 ; a[8..15] = K 32kt+8h+16..23  (two float4 loads each)
//  B (32x16 bf16): lane l -> N = l&15; same 16 K as A, packed 2/VGPR.
//  D (16x16 f32): VGPR r -> M = r + 8*(l>>4); N = l&15.
// ---------------------------------------------------------------------------
template <bool RELU>
__global__ __launch_bounds__(256) void gcn_gemm_bf16(
    const float* __restrict__ A,    // [N,128] f32 row-major
    const __bf16* __restrict__ Wb,  // [128,128] bf16 row-major (K major, N contig)
    float* __restrict__ H)          // [N,128] f32 out
{
    __shared__ __bf16 Wl[DIM * DIM];   // 32 KB; only shared object -> LDS offset 0

#if GCN_USE_TDM
    // ---- TDM: DMA the whole 32KB weight tile into LDS (wave 0 issues) ----
    if (threadIdx.x < 32) {
        const unsigned long long ga = (unsigned long long)(uintptr_t)Wb;
        // D# group 0: count=1 | lds_addr | global_addr[56:0] | type=2
        v4u g0 = { 1u,
                   0u,                                   // LDS byte address of Wl
                   (unsigned int)ga,
                   ((unsigned int)(ga >> 32) & 0x01FFFFFFu) | (2u << 30) };
        // D# group 1: wg_mask=0, data_size=1(2B), dims: tensor_dim0=16384,
        // tensor_dim1=1, tile_dim0=16384, tile_dim1=1, tensor_dim0_stride=16384
        v8i g1 = { (int)0x00010000,       // [17:16] data_size = 1 (2 bytes)
                   (int)0x40000000,       // [63:48] tensor_dim0 lo = 16384
                   (int)0x00010000,       // [95:80] tensor_dim1 lo = 1
                   (int)0x40000000,       // [127:112] tile_dim0 = 16384
                   (int)0x00000001,       // [143:128] tile_dim1 = 1, tile_dim2 = 0
                   (int)16384,            // [191:160] tensor_dim0_stride lo
                   0, 0 };
        v4i g2 = { 0, 0, 0, 0 };          // <=2D: dims 2..4 unused
        v4i g3 = { 0, 0, 0, 0 };
        v8i g4 = { 0, 0, 0, 0, 0, 0, 0, 0 };
        __builtin_amdgcn_tensor_load_to_lds(g0, g1, g2, g3, g4, 0);
        __builtin_amdgcn_s_wait_tensorcnt(0);
    }
#else
    // ---- fallback: cooperative stage of W into LDS (2048 x uint4) ----
    {
        const uint4* src = (const uint4*)Wb;
        uint4* dst = (uint4*)Wl;
#pragma unroll
        for (int i = 0; i < 8; ++i)
            dst[threadIdx.x + i * 256] = src[threadIdx.x + i * 256];
    }
#endif
    __syncthreads();

    const int wave = threadIdx.x >> 5;
    const int lane = threadIdx.x & 31;
    const int tile = blockIdx.x * (blockDim.x >> 5) + wave;
    if (tile >= N_TILES) return;

    const int m    = lane & 15;
    const int half = lane >> 4;
    const float* arow = A + (size_t)(tile * 16 + m) * DIM;

    v8f acc[8] = {};

#pragma unroll
    for (int kt = 0; kt < 4; ++kt) {
        // ---- load + convert A fragment: 4 x float4 (K-pairs are contiguous) ----
        const int kb = 32 * kt + 8 * half;
        float4 f0 = *(const float4*)(arow + kb + 0);
        float4 f1 = *(const float4*)(arow + kb + 4);
        float4 f2 = *(const float4*)(arow + kb + 16);
        float4 f3 = *(const float4*)(arow + kb + 20);
        if (RELU) {
            f0.x = fmaxf(f0.x, 0.f); f0.y = fmaxf(f0.y, 0.f); f0.z = fmaxf(f0.z, 0.f); f0.w = fmaxf(f0.w, 0.f);
            f1.x = fmaxf(f1.x, 0.f); f1.y = fmaxf(f1.y, 0.f); f1.z = fmaxf(f1.z, 0.f); f1.w = fmaxf(f1.w, 0.f);
            f2.x = fmaxf(f2.x, 0.f); f2.y = fmaxf(f2.y, 0.f); f2.z = fmaxf(f2.z, 0.f); f2.w = fmaxf(f2.w, 0.f);
            f3.x = fmaxf(f3.x, 0.f); f3.y = fmaxf(f3.y, 0.f); f3.z = fmaxf(f3.z, 0.f); f3.w = fmaxf(f3.w, 0.f);
        }
        v16bf a;
        a[0]  = (__bf16)f0.x; a[1]  = (__bf16)f0.y; a[2]  = (__bf16)f0.z; a[3]  = (__bf16)f0.w;
        a[4]  = (__bf16)f1.x; a[5]  = (__bf16)f1.y; a[6]  = (__bf16)f1.z; a[7]  = (__bf16)f1.w;
        a[8]  = (__bf16)f2.x; a[9]  = (__bf16)f2.y; a[10] = (__bf16)f2.z; a[11] = (__bf16)f2.w;
        a[12] = (__bf16)f3.x; a[13] = (__bf16)f3.y; a[14] = (__bf16)f3.z; a[15] = (__bf16)f3.w;

        // ---- B from LDS: load all 8 fragments first so ds_loads can issue
        //      ahead of the WMMA chain (partial dscnt waits) ----
        const int kg = kt * 32 + 8 * half;
        const __bf16* bcol = Wl + (lane & 15);
        v16bf b[8];
#pragma unroll
        for (int nt = 0; nt < 8; ++nt)
#pragma unroll
            for (int s = 0; s < 16; ++s)
                b[nt][s] = bcol[(kg + (s < 8 ? s : s + 8)) * DIM + nt * 16];
#pragma unroll
        for (int nt = 0; nt < 8; ++nt) {
            acc[nt] = __builtin_amdgcn_wmma_f32_16x16x32_bf16(
                false, a, false, b[nt], (short)0, acc[nt], false, false);
        }
    }

    // ---- store D (lanes 0-15 / 16-31 each write 64B-contiguous runs) ----
    float* hbase = H + (size_t)tile * 16 * DIM;
#pragma unroll
    for (int nt = 0; nt < 8; ++nt) {
        const int nn = nt * 16 + (lane & 15);
#pragma unroll
        for (int r = 0; r < 8; ++r) {
            const int mm = r + 8 * half;
            hbase[mm * DIM + nn] = acc[nt][r];
        }
    }
}

// ---------------------------------------------------------------------------
// SPMM: Hout[r] += vals[e] * Hin[cols[e]]  (atomic scatter)
// One wave per edge iteration: 32 lanes x float4 = 128 features exactly.
// Gathers and atomics stay L2-resident (H = 51.2MB << 192MB L2); the next
// edge's feature row is prefetched (global_prefetch_b8) to hide L2 latency.
// ---------------------------------------------------------------------------
__global__ __launch_bounds__(256) void gcn_spmm_atomic(
    const int* __restrict__ rows, const int* __restrict__ cols,
    const float* __restrict__ vals, const float* __restrict__ Hin,
    float* __restrict__ Hout)
{
    const int lane   = threadIdx.x & 31;
    const int wave   = (blockIdx.x * blockDim.x + threadIdx.x) >> 5;
    const int nWaves = (gridDim.x * blockDim.x) >> 5;

    for (int e = wave; e < N_EDGES; e += nWaves) {
        // prefetch next iteration's gather row into cache
        const int eNext = e + nWaves;
        if (eNext < N_EDGES) {
            const int cn = cols[eNext];
            __builtin_prefetch(Hin + (size_t)cn * DIM + lane * 4, 0, 3);
        }

        const int   r = __builtin_amdgcn_readfirstlane(rows[e]);
        const int   c = __builtin_amdgcn_readfirstlane(cols[e]);
        const float v = vals[e];

        const float4 x = *(const float4*)(Hin + (size_t)c * DIM + lane * 4);
        float* dst = Hout + (size_t)r * DIM + lane * 4;
        unsafeAtomicAdd(dst + 0, v * x.x);
        unsafeAtomicAdd(dst + 1, v * x.y);
        unsafeAtomicAdd(dst + 2, v * x.z);
        unsafeAtomicAdd(dst + 3, v * x.w);
    }
}

// ---------------------------------------------------------------------------
// Elementwise relu in place
// ---------------------------------------------------------------------------
__global__ __launch_bounds__(256) void gcn_relu(float* __restrict__ X, int n)
{
    int idx = blockIdx.x * blockDim.x + threadIdx.x;
    if (idx < n) X[idx] = fmaxf(X[idx], 0.0f);
}

// ---------------------------------------------------------------------------
extern "C" void kernel_launch(void* const* d_in, const int* in_sizes, int n_in,
                              void* d_out, int out_size, void* d_ws, size_t ws_size,
                              hipStream_t stream)
{
    const float* X    = (const float*)d_in[0];
    const int*   rows = (const int*)d_in[1];
    const int*   cols = (const int*)d_in[2];
    const float* vals = (const float*)d_in[3];
    const float* W1   = (const float*)d_in[4];
    const float* W2   = (const float*)d_in[5];
    float* out = (float*)d_out;

    __bf16* W1b = (__bf16*)d_ws;
    __bf16* W2b = W1b + DIM * DIM;
    float*  bufA = (float*)((char*)d_ws + 64 * 1024);

    const size_t hBytes = (size_t)N_NODES * DIM * sizeof(float);

    // 1. weights -> bf16
    gcn_convert_w<<<(DIM * DIM + 255) / 256, 256, 0, stream>>>(W1, W2, W1b, W2b);

    // 2. GEMM1: bufA = X @ W1
    const int gemmBlocks = (N_TILES + 7) / 8;  // 8 waves/block
    gcn_gemm_bf16<false><<<gemmBlocks, 256, 0, stream>>>(X, W1b, bufA);

    // 3. SPMM1 into d_out (pre-relu)
    (void)hipMemsetAsync(d_out, 0, hBytes, stream);
    gcn_spmm_atomic<<<2048, 256, 0, stream>>>(rows, cols, vals, bufA, out);

    // 4. GEMM2: bufA = relu(d_out) @ W2   (relu fused into A-load)
    gcn_gemm_bf16<true><<<gemmBlocks, 256, 0, stream>>>(out, W2b, bufA);

    // 5. SPMM2 into d_out
    (void)hipMemsetAsync(d_out, 0, hBytes, stream);
    gcn_spmm_atomic<<<2048, 256, 0, stream>>>(rows, cols, vals, bufA, out);

    // 6. final relu
    gcn_relu<<<(N_NODES * DIM + 255) / 256, 256, 0, stream>>>(out, N_NODES * DIM);
}
